// MixedAttnHeadEmbed_82076825027210
// MI455X (gfx1250) — matrix operand order repaired and checked by reading, not compiled, not run on hardware
//
#include <hip/hip_runtime.h>
#include <hip/hip_bf16.h>

typedef __bf16 bf16;
typedef __attribute__((ext_vector_type(16))) __bf16 v16bf;
typedef __attribute__((ext_vector_type(8)))  float  v8f;
typedef unsigned int u32x4_base __attribute__((ext_vector_type(4)));
typedef u32x4_base __attribute__((may_alias)) u32x4;

typedef unsigned int uintx4 __attribute__((ext_vector_type(4)));
typedef int          intx4  __attribute__((ext_vector_type(4)));
typedef int          intx8  __attribute__((ext_vector_type(8)));

#define BB 8
#define TT 1024
#define CC 768

#if __has_builtin(__builtin_amdgcn_tensor_load_to_lds) && __has_builtin(__builtin_amdgcn_s_wait_tensorcnt)
#define HAVE_TDM 1
#else
#define HAVE_TDM 0
#endif

// Load a 16-half WMMA fragment as two 16-byte chunks (LDS or global).
__device__ __forceinline__ v16bf ld_frag(const bf16* base, int o0, int o1) {
    union { v16bf v; u32x4 q[2]; } u;
    u.q[0] = *reinterpret_cast<const u32x4*>(base + o0);
    u.q[1] = *reinterpret_cast<const u32x4*>(base + o1);
    return u.v;
}

__device__ __forceinline__ bf16 to_bf16(float f) { return (bf16)f; }

#if HAVE_TDM
// TDM: DMA a 2-D tile [rows x rowlen] of bf16 from global into LDS (row-major,
// contiguous). D# layout per CDNA5 ISA ch.8 (groups 0..3). This toolchain's
// builtin takes 6 args: (g0 v4u32, g1 v8i32, g2 v4i32, g3 v4i32, v8i32, cpol).
__device__ __forceinline__ void tdm_load_tile_2d(const bf16* gsrc, bf16* lds_dst,
                                                 int rows, int rowlen) {
    unsigned long long ga = (unsigned long long)(const void*)gsrc;
    unsigned lds = (unsigned)(unsigned long long)(const void*)lds_dst; // low 32b = LDS offset
    unsigned rl = (unsigned)rowlen, rw = (unsigned)rows;

    uintx4 g0;
    g0[0] = 1u;                                              // count=1
    g0[1] = lds;                                             // lds_addr
    g0[2] = (unsigned)(ga & 0xFFFFFFFFu);                    // global_addr[31:0]
    g0[3] = (unsigned)((ga >> 32) & 0x01FFFFFFu) | (2u << 30); // addr[56:32] | type=2

    intx8 g1;
    g1[0] = (int)(1u << 16);                                 // data_size=1 (2 bytes)
    g1[1] = (int)((rl & 0xFFFFu) << 16);                     // tensor_dim0[15:0] @ bit48
    g1[2] = (int)((rl >> 16) | ((rw & 0xFFFFu) << 16));      // dim0 hi | tensor_dim1 lo
    g1[3] = (int)(((rw >> 16) & 0xFFFFu) | ((rl & 0xFFFFu) << 16)); // dim1 hi | tile_dim0
    g1[4] = (int)(rw & 0xFFFFu);                             // tile_dim1 | tile_dim2=0
    g1[5] = (int)rl;                                         // tensor_dim0_stride[31:0]
    g1[6] = 0;                                               // stride hi | dim1_stride lo
    g1[7] = 0;

    intx4 g2; g2[0] = 1; g2[1] = 1; g2[2] = 0; g2[3] = 0;    // tensor_dim2=1, dim3=1
    intx4 g3; g3[0] = 0; g3[1] = (int)(1u << 16); g3[2] = 0; g3[3] = 0; // tensor_dim4=1
    intx8 g4 = {0, 0, 0, 0, 0, 0, 0, 0};                     // unused (zero-filled)

    __builtin_amdgcn_tensor_load_to_lds(g0, g1, g2, g3, g4, 0);
}
#endif

// ---------------------------------------------------------------------------
__global__ void init_out_kernel(float* __restrict__ out, int n) {
    int i = blockIdx.x * blockDim.x + threadIdx.x;
    if (i < n) out[i] = 0.0f;
}

// ---------------------------------------------------------------------------
// Prep: build mixed Q/K/V for all 3 configs, bf16, layout [B, H, T, D].
// blockIdx.y = config (0:h=4, 1:h=8, 2:h=12)
// ---------------------------------------------------------------------------
__global__ void prep_kernel(const float* __restrict__ x,
                            const float* __restrict__ w,
                            bf16* __restrict__ ws) {
    const int cfg = blockIdx.y;
    const int h   = (cfg == 0) ? 4 : (cfg == 1) ? 8 : 12;
    const int D   = CC / h;
    const size_t total = (size_t)BB * TT * CC;
    size_t i = (size_t)blockIdx.x * blockDim.x + threadIdx.x;
    if (i >= total) return;

    const int col  = (int)(i % CC);
    const int t    = (int)((i / CC) % TT);
    const int b    = (int)(i / ((size_t)CC * TT));
    const int head = col / D;
    const int d    = col - head * D;
    const int hd0  = 384 / h;
    const int hd1  = 576 / h;

    const float w0 = w[cfg * 3 + 0];
    const float w1 = w[cfg * 3 + 1];
    const float w2 = w[cfg * 3 + 2];

    const float* row = x + ((size_t)b * TT + t) * (3 * CC);
    bf16* Q = ws + (size_t)cfg * 3 * total;
    bf16* K = Q + total;
    bf16* V = K + total;
    const size_t oidx = (((size_t)b * h + head) * TT + t) * (size_t)D + d;

    #pragma unroll
    for (int tn = 0; tn < 3; ++tn) {
        const float* src = row + tn * CC;
        float acc = w2 * src[head * D + d];
        if (d < hd0) acc += w0 * src[head * hd0 + d];
        if (d < hd1) acc += w1 * src[head * hd1 + d];
        bf16* dst = (tn == 0) ? Q : (tn == 1) ? K : V;
        dst[oidx] = to_bf16(acc);
    }
}

// ---------------------------------------------------------------------------
// Flash attention per config, bf16 WMMA, fp32 accumulate.
// Block: 256 threads = 8 waves; each wave owns 16 q-rows (block = 128 rows).
// ---------------------------------------------------------------------------
template<int D, int H>
__global__ void __launch_bounds__(256)
attn_kernel(const bf16* __restrict__ Qg, const bf16* __restrict__ Kg,
            const bf16* __restrict__ Vg, float* __restrict__ out) {
    constexpr int KCH = D / 32;
    constexpr int NDT = D / 16;

    __shared__ __align__(16) bf16 sK[32 * D];          // [token][d] row-major
    __shared__ __align__(16) bf16 sVt[D * 32];         // [d][token] (V transposed)
    __shared__ __align__(16) bf16 sP[8][16 * 32];      // per-wave P scratch [row][k]

    const int tid  = threadIdx.x;
    const int wave = tid >> 5;
    const int lane = tid & 31;
    const int g    = lane >> 4;
    const int ln   = lane & 15;
    const int head = blockIdx.y;
    const int b    = blockIdx.z;
    const int bh   = b * H + head;
    const int qBase = blockIdx.x * 128;
    const int qw0   = qBase + wave * 16;
    const float scale = 1.0f / sqrtf((float)D);

    const bf16* Qp = Qg + (size_t)bh * TT * D;
    const bf16* Kp = Kg + (size_t)bh * TT * D;
    const bf16* Vp = Vg + (size_t)bh * TT * D;

    v16bf qf[KCH];
    {
        const bf16* qrow = Qp + (size_t)(qw0 + ln) * D;
        #pragma unroll
        for (int c = 0; c < KCH; ++c)
            qf[c] = ld_frag(qrow, c * 32 + g * 8, c * 32 + 16 + g * 8);
    }

    v8f O[NDT];
    float m[8], l[8];
    #pragma unroll
    for (int t2 = 0; t2 < NDT; ++t2)
        #pragma unroll
        for (int j = 0; j < 8; ++j) O[t2][j] = 0.0f;
    #pragma unroll
    for (int j = 0; j < 8; ++j) { m[j] = -1e30f; l[j] = 0.0f; }

    const int kEnd  = qBase + 127;
    const int wqmax = qw0 + 15;

    for (int kc = 0; kc <= kEnd; kc += 32) {
        // ---- stage K tile: TDM DMA (wave 0) or cooperative copy ----
#if HAVE_TDM
        if (wave == 0) {
            tdm_load_tile_2d(Kp + (size_t)kc * D, sK, 32, D);
            __builtin_amdgcn_s_wait_tensorcnt(0);
            asm volatile("" ::: "memory");
        }
#else
        {
            const u32x4* src = reinterpret_cast<const u32x4*>(Kp + (size_t)kc * D);
            u32x4* dst = reinterpret_cast<u32x4*>(sK);
            for (int i = tid; i < 4 * D; i += 256) dst[i] = src[i];
        }
#endif
        // ---- stage V tile transposed (all waves) ----
        {
            const unsigned int* src =
                reinterpret_cast<const unsigned int*>(Vp + (size_t)kc * D);
            for (int i = tid; i < 16 * D; i += 256) {
                unsigned int u = src[i];
                int e2  = i * 2;
                int tok = e2 / D;
                int d0  = e2 - tok * D;
                sVt[d0 * 32 + tok]       = __builtin_bit_cast(bf16, (unsigned short)(u & 0xFFFFu));
                sVt[(d0 + 1) * 32 + tok] = __builtin_bit_cast(bf16, (unsigned short)(u >> 16));
            }
        }
        if (kc + 32 <= kEnd)
            __builtin_prefetch(Kp + (size_t)(kc + 32) * D, 0, 0);
        __syncthreads();

        if (kc <= wqmax) {   // wave-uniform: skip fully masked chunks
            // ---- S = Q K^T (two 16-col tiles), double-buffered B-frags ----
            v8f sAcc[2];
            #pragma unroll
            for (int nt = 0; nt < 2; ++nt) {
                v8f a;
                #pragma unroll
                for (int j = 0; j < 8; ++j) a[j] = 0.0f;
                const bf16* krow = sK + (nt * 16 + ln) * D;
                v16bf kcur = ld_frag(krow, g * 8, 16 + g * 8);
                #pragma unroll
                for (int c = 0; c < KCH; ++c) {
                    v16bf knext = kcur;
                    if (c + 1 < KCH)
                        knext = ld_frag(krow, (c + 1) * 32 + g * 8,
                                              (c + 1) * 32 + 16 + g * 8);
                    a = __builtin_amdgcn_wmma_f32_16x16x32_bf16(
                            false, qf[c], false, kcur, (short)0, a, false, false);
                    kcur = knext;
                }
                #pragma unroll
                for (int j = 0; j < 8; ++j) {
                    float s = a[j] * scale;
                    int qg = qw0 + j + 8 * g;
                    int kg = kc + nt * 16 + ln;
                    if (kg > qg) s = -1e30f;
                    a[j] = s;
                }
                sAcc[nt] = a;
            }
            // ---- online softmax ----
            float mn[8], corr[8];
            #pragma unroll
            for (int j = 0; j < 8; ++j) {
                float rm = fmaxf(sAcc[0][j], sAcc[1][j]);
                rm = fmaxf(rm, __shfl_xor(rm, 1, 32));
                rm = fmaxf(rm, __shfl_xor(rm, 2, 32));
                rm = fmaxf(rm, __shfl_xor(rm, 4, 32));
                rm = fmaxf(rm, __shfl_xor(rm, 8, 32));
                mn[j]   = fmaxf(m[j], rm);
                corr[j] = __expf(m[j] - mn[j]);
                m[j]    = mn[j];
            }
            bf16* pw = &sP[wave][0];
            float rs[8];
            #pragma unroll
            for (int j = 0; j < 8; ++j) {
                float p0 = __expf(sAcc[0][j] - mn[j]);
                float p1 = __expf(sAcc[1][j] - mn[j]);
                rs[j] = p0 + p1;
                int r = (j + 8 * g) * 32;
                pw[r + ln]      = to_bf16(p0);
                pw[r + 16 + ln] = to_bf16(p1);
            }
            #pragma unroll
            for (int j = 0; j < 8; ++j) {
                float s = rs[j];
                s += __shfl_xor(s, 1, 32);
                s += __shfl_xor(s, 2, 32);
                s += __shfl_xor(s, 4, 32);
                s += __shfl_xor(s, 8, 32);
                l[j] = l[j] * corr[j] + s;
            }
            #pragma unroll
            for (int t2 = 0; t2 < NDT; ++t2)
                #pragma unroll
                for (int j = 0; j < 8; ++j) O[t2][j] *= corr[j];

            asm volatile("" ::: "memory");   // keep P stores before A-frag read
            // ---- P x V, double-buffered V B-frags ----
            v16bf pf = ld_frag(pw, ln * 32 + g * 8, ln * 32 + 16 + g * 8);
            v16bf vcur = ld_frag(sVt, ln * 32 + g * 8, ln * 32 + 16 + g * 8);
            #pragma unroll
            for (int t2 = 0; t2 < NDT; ++t2) {
                v16bf vnext = vcur;
                if (t2 + 1 < NDT)
                    vnext = ld_frag(sVt, ((t2 + 1) * 16 + ln) * 32 + g * 8,
                                         ((t2 + 1) * 16 + ln) * 32 + 16 + g * 8);
                O[t2] = __builtin_amdgcn_wmma_f32_16x16x32_bf16(
                            false, pf, false, vcur, (short)0, O[t2], false, false);
                vcur = vnext;
            }
        }
        __syncthreads();
    }

    // ---- epilogue: out[b, q, head*D + col] += O / l ----
    #pragma unroll
    for (int t2 = 0; t2 < NDT; ++t2) {
        #pragma unroll
        for (int j = 0; j < 8; ++j) {
            float inv = 1.0f / l[j];
            size_t idx = ((size_t)b * TT + (qw0 + j + 8 * g)) * CC
                       + head * D + t2 * 16 + ln;
            out[idx] += O[t2][j] * inv;
        }
    }
}

// ---------------------------------------------------------------------------
extern "C" void kernel_launch(void* const* d_in, const int* in_sizes, int n_in,
                              void* d_out, int out_size, void* d_ws, size_t ws_size,
                              hipStream_t stream) {
    const float* x = (const float*)d_in[0];     // [8,1024,2304]
    const float* w = (const float*)d_in[1];     // [9]
    float* out = (float*)d_out;                 // [8,1024,768]

    const size_t S = (size_t)BB * TT * CC;
    bf16* ws = (bf16*)d_ws;                     // 9 * S bf16 = ~113 MB

    {
        int n = (int)S;
        init_out_kernel<<<(n + 255) / 256, 256, 0, stream>>>(out, n);
    }
    {
        dim3 grid((unsigned)((S + 255) / 256), 3);
        prep_kernel<<<grid, 256, 0, stream>>>(x, w, ws);
    }
    {
        bf16* Q0 = ws + 0 * 3 * S; bf16* K0 = Q0 + S; bf16* V0 = K0 + S;
        bf16* Q1 = ws + 1 * 3 * S; bf16* K1 = Q1 + S; bf16* V1 = K1 + S;
        bf16* Q2 = ws + 2 * 3 * S; bf16* K2 = Q2 + S; bf16* V2 = K2 + S;
        attn_kernel<192, 4><<<dim3(TT / 128,  4, BB), 256, 0, stream>>>(Q0, K0, V0, out);
        attn_kernel< 96, 8><<<dim3(TT / 128,  8, BB), 256, 0, stream>>>(Q1, K1, V1, out);
        attn_kernel< 64,12><<<dim3(TT / 128, 12, BB), 256, 0, stream>>>(Q2, K2, V2, out);
    }
}
